// PointNetSetAbstraction_14826227106354
// MI455X (gfx1250) — compile-verified, matrix operand order checked by reading, not run on hardware
//
#include <hip/hip_runtime.h>
#include <hip/hip_bf16.h>

typedef __attribute__((ext_vector_type(16))) _Float16 v16h;
typedef __attribute__((ext_vector_type(8)))  float    v8f;

#define BATCH   16
#define NPTS    4096
#define SCENT   1024
#define NSAMP   32
#define NCENT   (BATCH * SCENT)          // 16384
#define MROWS   (NCENT * NSAMP)          // 524288
#define RAD2    (0.2f * 0.2f)
#define BN_EPS  1e-5f

// ---------------------------------------------------------------- utilities
__global__ void zero_stats_kernel(float* stats) {
    int i = threadIdx.x;
    for (int k = i; k < 768; k += 256) stats[k] = 0.f;
}

// Pack W0 (64x9 -> 64x32 padded), W1 (64x64), W2 (128x64) to f16.
__global__ void pack_weights_kernel(const float* __restrict__ W0,
                                    const float* __restrict__ W1,
                                    const float* __restrict__ W2,
                                    _Float16* __restrict__ Wp) {
    int i = blockIdx.x * 256 + threadIdx.x;
    const int n0 = 64 * 32, n1 = 64 * 64, n2 = 128 * 64;
    if (i < n0) {
        int o = i >> 5, k = i & 31;
        Wp[i] = (_Float16)((k < 9) ? W0[o * 9 + k] : 0.f);
    } else if (i < n0 + n1) {
        int j = i - n0;
        Wp[i] = (_Float16)W1[j];
    } else if (i < n0 + n1 + n2) {
        int j = i - n0 - n1;
        Wp[i] = (_Float16)W2[j];
    }
}

// ---------------------------------------------------------------- FPS
// One block per batch. Each thread owns 16 points (p = i*256 + t), keeps its
// running min-distance in registers, block argmax in LDS each step.
__global__ void fps_kernel(const float* __restrict__ xyz,
                           float* __restrict__ new_xyz /* NCENT x 3 */) {
    int b = blockIdx.x, t = threadIdx.x;
    const float* xb = xyz + (size_t)b * NPTS * 3;
    float dist[16];
#pragma unroll
    for (int i = 0; i < 16; ++i) dist[i] = 1e10f;

    __shared__ float s_c[3];
    __shared__ float s_val[256];
    __shared__ int   s_idx[256];
    __shared__ int   s_far;
    if (t == 0) s_far = 0;
    __syncthreads();

    for (int it = 0; it < SCENT; ++it) {
        if (t == 0) {
            int far = s_far;
            float cx = xb[far * 3 + 0], cy = xb[far * 3 + 1], cz = xb[far * 3 + 2];
            float* o = new_xyz + ((size_t)b * SCENT + it) * 3;
            o[0] = cx; o[1] = cy; o[2] = cz;
            s_c[0] = cx; s_c[1] = cy; s_c[2] = cz;
        }
        __syncthreads();
        float cx = s_c[0], cy = s_c[1], cz = s_c[2];
        float bestv = -1.f; int besti = 0;
#pragma unroll
        for (int i = 0; i < 16; ++i) {
            int p = i * 256 + t;
            float dx = xb[p * 3 + 0] - cx;
            float dy = xb[p * 3 + 1] - cy;
            float dz = xb[p * 3 + 2] - cz;
            float d = dx * dx + dy * dy + dz * dz;
            if (d < dist[i]) dist[i] = d;
            if (dist[i] > bestv) { bestv = dist[i]; besti = p; }
        }
        s_val[t] = bestv; s_idx[t] = besti;
        __syncthreads();
        for (int s = 128; s > 0; s >>= 1) {
            if (t < s) {
                float v2 = s_val[t + s]; int i2 = s_idx[t + s];
                if (v2 > s_val[t] || (v2 == s_val[t] && i2 < s_idx[t])) {
                    s_val[t] = v2; s_idx[t] = i2;
                }
            }
            __syncthreads();
        }
        if (t == 0) s_far = s_idx[0];
        __syncthreads();
    }
}

// ---------------------------------------------------------------- ball query
// One wave32 per center: scan points in index order, ballot + prefix popcount
// selects the first 32 in-radius indices; remaining slots filled with first.
__global__ void ballq_kernel(const float* __restrict__ xyz,
                             const float* __restrict__ new_xyz,
                             int* __restrict__ idxbuf) {
    int wave = threadIdx.x >> 5, lane = threadIdx.x & 31;
    int center = blockIdx.x * 8 + wave;
    int b = center >> 10;
    const float* xb = xyz + (size_t)b * NPTS * 3;
    float cx = new_xyz[center * 3 + 0];
    float cy = new_xyz[center * 3 + 1];
    float cz = new_xyz[center * 3 + 2];

    int cnt = 0, first = -1;
    for (int base = 0; base < NPTS && cnt < NSAMP; base += 32) {
        int p = base + lane;
        float dx = xb[p * 3 + 0] - cx;
        float dy = xb[p * 3 + 1] - cy;
        float dz = xb[p * 3 + 2] - cz;
        bool in = (dx * dx + dy * dy + dz * dz) <= RAD2;
        unsigned mask = (unsigned)__ballot(in);
        if (first < 0 && mask) first = base + __builtin_ctz(mask);
        int pos = cnt + __builtin_popcount(mask & ((1u << lane) - 1u));
        if (in && pos < NSAMP) idxbuf[(size_t)center * NSAMP + pos] = p;
        cnt += __builtin_popcount(mask);
    }
    if (cnt > NSAMP) cnt = NSAMP;
    if (first < 0) first = 0;
    if (lane >= cnt) idxbuf[(size_t)center * NSAMP + lane] = first;
}

// ---------------------------------------------------------------- grouping
// Build X0 (MROWS x 32 f16): [xyz - center (3), points (6), zero pad (23)]
__global__ void group_pack_kernel(const float* __restrict__ xyz,
                                  const float* __restrict__ points,
                                  const float* __restrict__ new_xyz,
                                  const int* __restrict__ idxbuf,
                                  _Float16* __restrict__ X0) {
    int i = blockIdx.x * 256 + threadIdx.x;   // over MROWS
    if (i >= MROWS) return;
    int center = i >> 5;
    int b = center >> 10;
    int p = idxbuf[i];
    const float* xp = xyz    + ((size_t)b * NPTS + p) * 3;
    const float* pp = points + ((size_t)b * NPTS + p) * 6;
    const float* c  = new_xyz + (size_t)center * 3;
    _Float16* xo = X0 + (size_t)i * 32;
    xo[0] = (_Float16)(xp[0] - c[0]);
    xo[1] = (_Float16)(xp[1] - c[1]);
    xo[2] = (_Float16)(xp[2] - c[2]);
#pragma unroll
    for (int k = 0; k < 6; ++k) xo[3 + k] = (_Float16)pp[k];
#pragma unroll
    for (int k = 9; k < 32; ++k) xo[k] = (_Float16)0.f;
}

// ---------------------------------------------------------------- WMMA GEMM
// H[M x Nc] = X[M x K] @ Wp^T + bias, Wp is (Nc x K) row-major (== B matrix).
// One 16x16 output tile per wave via v_wmma_f32_16x16x32_f16; per-channel
// sum / sumsq accumulated for the global batchnorm.
__global__ void gemm_wmma_kernel(const _Float16* __restrict__ X,
                                 const _Float16* __restrict__ Wp,
                                 const float* __restrict__ bias,
                                 _Float16* __restrict__ H,
                                 float* __restrict__ stats,  // [2*Nc]
                                 int K, int Nc) {
    __shared__ float s_sum[128];
    __shared__ float s_sq[128];
    for (int c = threadIdx.x; c < 128; c += 256) { s_sum[c] = 0.f; s_sq[c] = 0.f; }
    __syncthreads();

    int wave = threadIdx.x >> 5, lane = threadIdx.x & 31;
    int tile = blockIdx.x * 8 + wave;
    int ntn = Nc >> 4;
    int tm = tile / ntn, tn = tile % ntn;

    int row   = tm * 16 + (lane & 15);
    int col   = tn * 16 + (lane & 15);
    int kbase = (lane < 16) ? 0 : 8;

    v8f acc = {};
    for (int k0 = 0; k0 < K; k0 += 32) {
        v16h a, bm;
        const _Float16* xp = X  + (size_t)row * K + k0 + kbase;
        const _Float16* wp = Wp + (size_t)col * K + k0 + kbase;
#pragma unroll
        for (int h = 0; h < 8; ++h) {
            a[h]      = xp[h];
            a[8 + h]  = xp[16 + h];
            bm[h]     = wp[h];
            bm[8 + h] = wp[16 + h];
        }
        acc = __builtin_amdgcn_wmma_f32_16x16x32_f16(
            /*neg_a=*/false, a, /*neg_b=*/false, bm,
            /*c_mod=*/(short)0, acc, /*reuse_a=*/false, /*reuse_b=*/false);
    }

    float bcol = bias[col];
    float lsum = 0.f, lsq = 0.f;
#pragma unroll
    for (int r = 0; r < 8; ++r) {
        float v = acc[r] + bcol;
        int orow = tm * 16 + r + ((lane < 16) ? 0 : 8);
        H[(size_t)orow * Nc + col] = (_Float16)v;
        lsum += v; lsq += v * v;
    }
    // lanes l and l+16 hold the same column: pairwise reduce, then LDS.
    lsum += __shfl_down(lsum, 16);
    lsq  += __shfl_down(lsq, 16);
    if (lane < 16) {
        atomicAdd(&s_sum[col], lsum);
        atomicAdd(&s_sq[col],  lsq);
    }
    __syncthreads();
    if (threadIdx.x < (unsigned)Nc) {
        atomicAdd(&stats[threadIdx.x],      s_sum[threadIdx.x]);
        atomicAdd(&stats[Nc + threadIdx.x], s_sq[threadIdx.x]);
    }
}

// ---------------------------------------------------------------- BN + ReLU
__global__ void bn_relu_kernel(const _Float16* __restrict__ H,
                               const float* __restrict__ stats,
                               const float* __restrict__ g,
                               const float* __restrict__ bb,
                               _Float16* __restrict__ Xn, int Nc) {
    size_t i = (size_t)blockIdx.x * 256 + threadIdx.x;
    if (i >= (size_t)MROWS * Nc) return;
    int c = (int)(i % Nc);
    float invM = 1.f / (float)MROWS;
    float mu  = stats[c] * invM;
    float var = stats[Nc + c] * invM - mu * mu;
    float v = ((float)H[i] - mu) * rsqrtf(var + BN_EPS) * g[c] + bb[c];
    Xn[i] = (_Float16)fmaxf(v, 0.f);
}

// ------------------------------------------------ final BN + ReLU + max pool
__global__ void bn_relu_max_kernel(const _Float16* __restrict__ H,
                                   const float* __restrict__ stats,
                                   const float* __restrict__ g,
                                   const float* __restrict__ bb,
                                   float* __restrict__ out, int Nc) {
    int i = blockIdx.x * 256 + threadIdx.x;   // over NCENT * Nc
    if (i >= NCENT * Nc) return;
    int c = i % Nc, s = i / Nc;
    float invM = 1.f / (float)MROWS;
    float mu  = stats[c] * invM;
    float scale = rsqrtf(stats[Nc + c] * invM - mu * mu + BN_EPS) * g[c];
    float off = bb[c];
    float m = 0.f;   // relu output is >= 0, and max over 32 relu'd values
#pragma unroll 4
    for (int j = 0; j < NSAMP; ++j) {
        float v = ((float)H[((size_t)s * NSAMP + j) * Nc + c] - mu) * scale + off;
        v = fmaxf(v, 0.f);
        m = fmaxf(m, v);
    }
    out[(size_t)s * Nc + c] = m;
}

// ---------------------------------------------------------------- launch
extern "C" void kernel_launch(void* const* d_in, const int* in_sizes, int n_in,
                              void* d_out, int out_size, void* d_ws, size_t ws_size,
                              hipStream_t stream) {
    const float* xyz    = (const float*)d_in[0];
    const float* points = (const float*)d_in[1];
    const float* W0 = (const float*)d_in[2];
    const float* b0 = (const float*)d_in[3];
    const float* g0 = (const float*)d_in[4];
    const float* bb0 = (const float*)d_in[5];
    const float* W1 = (const float*)d_in[6];
    const float* b1 = (const float*)d_in[7];
    const float* g1 = (const float*)d_in[8];
    const float* bb1 = (const float*)d_in[9];
    const float* W2 = (const float*)d_in[10];
    const float* b2 = (const float*)d_in[11];
    const float* g2 = (const float*)d_in[12];
    const float* bb2 = (const float*)d_in[13];

    float* out = (float*)d_out;
    float* new_xyz    = out;                       // NCENT*3
    float* new_points = out + (size_t)NCENT * 3;   // NCENT*128

    // workspace layout
    char* ws = (char*)d_ws;
    int*      idxbuf = (int*)ws;                                     // 2 MB
    float*    stats  = (float*)(ws + 2 * 1024 * 1024);               // 3 sets of 256 f32
    _Float16* Wp     = (_Float16*)(ws + 2 * 1024 * 1024 + 4096);     // 14336 halves
    _Float16* Xbuf   = (_Float16*)(ws + 2 * 1024 * 1024 + 4096 + 32768);        // 64 MB
    _Float16* Hbuf   = (_Float16*)(ws + 2 * 1024 * 1024 + 4096 + 32768
                                      + (size_t)MROWS * 64 * 2);                // 128 MB

    _Float16* Wp0 = Wp;
    _Float16* Wp1 = Wp + 64 * 32;
    _Float16* Wp2 = Wp + 64 * 32 + 64 * 64;
    float* st0 = stats;
    float* st1 = stats + 256;
    float* st2 = stats + 512;

    zero_stats_kernel<<<1, 256, 0, stream>>>(stats);
    pack_weights_kernel<<<(64 * 32 + 64 * 64 + 128 * 64 + 255) / 256, 256, 0, stream>>>(
        W0, W1, W2, Wp);

    fps_kernel<<<BATCH, 256, 0, stream>>>(xyz, new_xyz);
    ballq_kernel<<<NCENT / 8, 256, 0, stream>>>(xyz, new_xyz, idxbuf);
    group_pack_kernel<<<MROWS / 256, 256, 0, stream>>>(xyz, points, new_xyz, idxbuf, Xbuf);

    // Layer 0: K=32 (padded from 9), Nc=64.  tiles = (M/16)*(64/16) = 131072 -> 16384 blocks
    gemm_wmma_kernel<<<16384, 256, 0, stream>>>(Xbuf, Wp0, b0, Hbuf, st0, 32, 64);
    bn_relu_kernel<<<(int)(((size_t)MROWS * 64 + 255) / 256), 256, 0, stream>>>(
        Hbuf, st0, g0, bb0, Xbuf, 64);

    // Layer 1: K=64, Nc=64
    gemm_wmma_kernel<<<16384, 256, 0, stream>>>(Xbuf, Wp1, b1, Hbuf, st1, 64, 64);
    bn_relu_kernel<<<(int)(((size_t)MROWS * 64 + 255) / 256), 256, 0, stream>>>(
        Hbuf, st1, g1, bb1, Xbuf, 64);

    // Layer 2: K=64, Nc=128. tiles = 32768*8 = 262144 -> 32768 blocks
    gemm_wmma_kernel<<<32768, 256, 0, stream>>>(Xbuf, Wp2, b2, Hbuf, st2, 64, 128);
    bn_relu_max_kernel<<<(NCENT * 128 + 255) / 256, 256, 0, stream>>>(
        Hbuf, st2, g2, bb2, new_points, 128);
}
